// ProgressiveGaussianSplatter_75033078661509
// MI455X (gfx1250) — matrix operand order repaired and checked by reading, not compiled, not run on hardware
//
#include <hip/hip_runtime.h>

// CDNA5 / gfx1250: wave32, WMMA 16x16x32 f16 -> f32 accumulate.
typedef __attribute__((ext_vector_type(16))) _Float16 v16h;
typedef __attribute__((ext_vector_type(8)))  float    v8f;

#define GS_N  1024          // gaussians (matches setup_inputs)
#define GS_W  256
#define GS_H  256
#define GS_PIX (GS_W * GS_H)

// exp(-0.5*t) == exp2(t * (-0.5 * log2(e)))
#define GS_NEG_HALF_LOG2E (-0.72134752044448170367996234050095f)

// Each wave: 16 pixels (A rows), loop K over gaussians in steps of 32.
// A (16x32 f16)  = gaussian weights  (computed in f32 VALU, cvt to f16)
// B (32x16 f16)  = alpha-premultiplied colors, column n = channel (n<3)
// C (16x16 f32)  = accumulated out[pixel, channel]
__global__ __launch_bounds__(256) void gsplat_wmma_kernel(
    const float* __restrict__ position,   // (N,2)
    const float* __restrict__ color,      // (N,3)
    const float* __restrict__ log_scale,  // (N,2)
    const float* __restrict__ rotation,   // (N)
    const float* __restrict__ alphas,     // (N)
    float* __restrict__ out,              // (3,H,W)
    int n_gauss)
{
    // geometry: px, py, cos, sin, 1/sx, 1/sy, pad, pad  (32B/gaussian -> b128+b64 loads)
    __shared__ __align__(16) float    s_geom[GS_N * 8];          // 32 KB
    __shared__ __align__(32) _Float16 s_colT[3 * GS_N];          // 6 KB, [c][k] transposed

    // ---- one-time per-block preprocessing into LDS ----------------------
    for (int k = threadIdx.x; k < n_gauss; k += blockDim.x) {
        const float sx = __expf(log_scale[2*k + 0]) + 1e-6f;
        const float sy = __expf(log_scale[2*k + 1]) + 1e-6f;
        const float r  = rotation[k];
        const float al = alphas[k];
        s_geom[8*k + 0] = position[2*k + 0];
        s_geom[8*k + 1] = position[2*k + 1];
        s_geom[8*k + 2] = __cosf(r);
        s_geom[8*k + 3] = __sinf(r);
        s_geom[8*k + 4] = 1.0f / sx;
        s_geom[8*k + 5] = 1.0f / sy;
        s_geom[8*k + 6] = 0.0f;
        s_geom[8*k + 7] = 0.0f;
        s_colT[0*GS_N + k] = (_Float16)(al * color[3*k + 0]);
        s_colT[1*GS_N + k] = (_Float16)(al * color[3*k + 1]);
        s_colT[2*GS_N + k] = (_Float16)(al * color[3*k + 2]);
    }
    __syncthreads();

    const int lane = threadIdx.x & 31;
    const int wave = threadIdx.x >> 5;
    const int pixel_base = (blockIdx.x * (blockDim.x >> 5) + wave) << 4;  // 16 px/wave

    // A 16x32 layout used: row M = lane%16; lanes<16 hold K 0..15,
    // lanes>=16 hold K 16..31.  B 32x16: col N = lane%16; lanes<16 K 0..15,
    // lanes>=16 K 16..31.
    const int m   = lane & 15;
    const int klo = (lane < 16) ? 0 : 16;   // A K-half offset
    const int khi = (lane < 16) ? 0 : 16;   // B K-half offset
    const int n   = lane & 15;              // output channel (valid if < 3)
    const int cc  = (n < 3) ? n : 0;        // clamp: cols 3..15 of D are never read

    // pixel coords for this lane's A row
    const int   p = pixel_base + m;
    const float x = -1.0f + (float)(p & (GS_W - 1)) * (2.0f / (GS_W - 1));
    const float y = -1.0f + (float)(p >> 8)         * (2.0f / (GS_H - 1));

    const _Float16* bsrc = &s_colT[cc * GS_N + khi];

    v8f acc = {};

    const int ksteps = n_gauss >> 5;        // 32 gaussians per WMMA
    #pragma unroll 1
    for (int kb = 0; kb < ksteps; ++kb) {
        const int g0 = kb << 5;

        // ---- B fragment first: issue ds_load_b128 x2 early so the LDS
        // latency hides behind the long A-fragment VALU/exp chain.
        // 32B aligned: cc*2048 + g0*2 (64B mult) + khi*2 (0 or 32)
        const v16h b = *(const v16h*)(bsrc + g0);

        // ---- A fragment: 16 gaussian weights for this lane's pixel ------
        v16h a;
        #pragma unroll
        for (int j = 0; j < 16; ++j) {
            const int jj = j;                           // K pair mapping
            const int k  = g0 + klo + jj;
            const float4 g  = *(const float4*)&s_geom[k * 8];       // px,py,cos,sin
            const float2 gi = *(const float2*)&s_geom[k * 8 + 4];   // 1/sx, 1/sy
            const float dx = x - g.x;
            const float dy = y - g.y;
            const float xr = g.z * dx + g.w * dy;
            const float yr = g.z * dy - g.w * dx;
            const float q1 = xr * gi.x;
            const float q2 = yr * gi.y;
            const float t  = q1 * q1 + q2 * q2;
            // single mul + native v_exp_f32
            a[j] = (_Float16)__builtin_amdgcn_exp2f(t * GS_NEG_HALF_LOG2E);
        }

        // D = A x B + C   (XDL op; co-executes with next iteration's VALU)
        acc = __builtin_amdgcn_wmma_f32_16x16x32_f16(
                  false, a, false, b, (short)0, acc, false, false);
    }

    // ---- writeback: D layout -> VGPR j holds M = j (+8 for lanes>=16) ---
    if (n < 3) {
        const int mhi = (lane < 16) ? 0 : 8;
        float* o = out + n * GS_PIX + pixel_base + mhi;
        const float4 lo = make_float4(acc[0] - 1.0f, acc[1] - 1.0f,
                                      acc[2] - 1.0f, acc[3] - 1.0f);
        const float4 hi = make_float4(acc[4] - 1.0f, acc[5] - 1.0f,
                                      acc[6] - 1.0f, acc[7] - 1.0f);
        *(float4*)(o + 0) = lo;     // global_store_b128
        *(float4*)(o + 4) = hi;     // global_store_b128
    }
}

extern "C" void kernel_launch(void* const* d_in, const int* in_sizes, int n_in,
                              void* d_out, int out_size, void* d_ws, size_t ws_size,
                              hipStream_t stream) {
    const float* position  = (const float*)d_in[0];
    const float* color     = (const float*)d_in[1];
    const float* log_scale = (const float*)d_in[2];
    const float* rotation  = (const float*)d_in[3];
    const float* alphas    = (const float*)d_in[4];
    float* out             = (float*)d_out;

    int n_gauss = in_sizes[3];          // rotation is (1,N) -> flat N
    if (n_gauss > GS_N) n_gauss = GS_N;

    const int block = 256;                              // 8 wave32 per block
    const int waves_total = GS_PIX / 16;                // 4096 waves
    const int grid = waves_total / (block / 32);        // 512 blocks

    gsplat_wmma_kernel<<<grid, block, 0, stream>>>(
        position, color, log_scale, rotation, alphas, out, n_gauss);
}
